// MHA_83674552861107
// MI455X (gfx1250) — compile-verified, matrix operand order checked by reading, not run on hardware
//
#include <hip/hip_runtime.h>
#include <hip/hip_bf16.h>

typedef __attribute__((ext_vector_type(16))) _Float16 v16h;
typedef __attribute__((ext_vector_type(8)))  _Float16 v8h;
typedef __attribute__((ext_vector_type(8)))  float    v8f;

#define B_  2
#define T_  2048
#define D_  1024
#define H_  16
#define HD_ 64
#define D3_ 3072
#define NT_ 4096   /* B_*T_ */

// ---------------------------------------------------------------------------
// Fragment loader for 16-bit WMMA A/B operands (wave32 layout, 16x16x32):
//   lanes 0-15  hold entity (row/col) = base+lane, K = {k0..k0+7, k0+16..k0+23}
//   lanes 16-31 hold entity           = base+lane-16, K = {k0+8..15, k0+24..31}
// With row-major storage of the entity's K-line this is two b128 loads.
// ---------------------------------------------------------------------------
__device__ __forceinline__ v16h ldfrag_g(const _Float16* p, int hi) {
  const v8h* q = (const v8h*)(p + (hi << 3));
  v8h lo = q[0];
  v8h hh = q[2];                       // +16 halves
  return __builtin_shufflevector(lo, hh, 0,1,2,3,4,5,6,7,8,9,10,11,12,13,14,15);
}

__device__ __forceinline__ v8f wmma_f16(v16h a, v16h b, v8f c) {
  return __builtin_amdgcn_wmma_f32_16x16x32_f16(false, a, false, b,
                                                (short)0, c, false, false);
}

// ------------------------------- converters --------------------------------
__global__ __launch_bounds__(256) void k_cvt(const float* __restrict__ src,
                                             _Float16* __restrict__ dst, int n) {
  int i = blockIdx.x * 256 + threadIdx.x;
  if (i < n) dst[i] = (_Float16)src[i];
}

// src [R][C] f32  ->  dst [C][R] f16 (transpose + cast); writes coalesced
__global__ __launch_bounds__(256) void k_cvt_t(const float* __restrict__ src,
                                               _Float16* __restrict__ dst,
                                               int R, int C) {
  int i = blockIdx.x * 256 + threadIdx.x;
  if (i < R * C) {
    int c = i / R, r = i - c * R;
    dst[i] = (_Float16)src[(size_t)r * C + c];
  }
}

// --------------------------- QKV projection GEMM ---------------------------
// [4096 x 1024] @ [1024 x 3072] + b1. Wave tile: 32 rows x 64 cols
// (2 A fragments amortize the 4 B fragments -> 1.5 b128 loads per WMMA).
// Output scattered directly into per-head Q [B,H,T,64], K [B,H,T,64],
// V-transposed [B,H,64,T] f16 buffers (attention-friendly layouts).
__global__ __launch_bounds__(256) void k_gemm_qkv(
    const _Float16* __restrict__ xh, const _Float16* __restrict__ w1t,
    const float* __restrict__ bias1,
    _Float16* __restrict__ qh, _Float16* __restrict__ kh,
    _Float16* __restrict__ vt) {
  const int lane = threadIdx.x & 31;
  const int wave = threadIdx.x >> 5;
  const int nl = lane & 15, hi = lane >> 4;

  const int n_base = blockIdx.x * 64;                 // 0..3071
  const int m_base = blockIdx.y * 256 + wave * 32;    // 0..4095

  const _Float16* arow0 = xh + (size_t)(m_base + nl) * D_;
  const _Float16* arow1 = xh + (size_t)(m_base + 16 + nl) * D_;
  const _Float16* brow  = w1t + (size_t)(n_base + nl) * D_;

  v8f c[2][4] = {};
  for (int k0 = 0; k0 < D_; k0 += 32) {
    v16h a0 = ldfrag_g(arow0 + k0, hi);
    v16h a1 = ldfrag_g(arow1 + k0, hi);
    __builtin_prefetch(arow0 + k0 + 256, 0, 1);
    __builtin_prefetch(arow1 + k0 + 256, 0, 1);
#pragma unroll
    for (int j = 0; j < 4; ++j) {
      v16h bj = ldfrag_g(brow + (size_t)j * 16 * D_ + k0, hi);
      c[0][j] = wmma_f16(a0, bj, c[0][j]);
      c[1][j] = wmma_f16(a1, bj, c[1][j]);
    }
  }

  // C layout: lanes 0-15 -> M=r, N=nl ; lanes 16-31 -> M=r+8, N=nl
  const int seg  = n_base >> 10;           // 0:q 1:k 2:v (uniform per block)
  const int head = (n_base & 1023) >> 6;   // uniform per block
#pragma unroll
  for (int g = 0; g < 2; ++g) {
#pragma unroll
    for (int j = 0; j < 4; ++j) {
      const int hd = j * 16 + nl;          // n_base % 64 == 0
      const float bv = bias1[n_base + hd];
#pragma unroll
      for (int r = 0; r < 8; ++r) {
        const int row = m_base + g * 16 + r + (hi << 3);
        const int bb = row >> 11;
        const int t  = row & (T_ - 1);
        const float v = c[g][j][r] + bv;
        if (seg == 0)
          qh[((size_t)(bb * H_ + head) * T_ + t) * HD_ + hd] = (_Float16)v;
        else if (seg == 1)
          kh[((size_t)(bb * H_ + head) * T_ + t) * HD_ + hd] = (_Float16)v;
        else
          vt[((size_t)(bb * H_ + head) * HD_ + hd) * T_ + t] = (_Float16)v;
      }
    }
  }
}

// ------------------------- flash attention (causal) ------------------------
// Grid (T/128, B*H); 8 waves/block cover 128 query rows of one head and walk
// 32-key tiles IN LOCKSTEP: the block cooperatively stages the K tile (4KB)
// and transposed-V tile (4KB) in LDS (one global b128 load per tile-element,
// shared by all 8 waves), then every wave builds its WMMA fragments with
// ds_load_b128. Online softmax keeps everything else in VGPRs; the T x T
// matrix never exists in memory.
__global__ __launch_bounds__(256) void k_attn(
    const _Float16* __restrict__ qh, const _Float16* __restrict__ kh,
    const _Float16* __restrict__ vt, _Float16* __restrict__ yh) {
  __shared__ __align__(16) _Float16 ksh[32][64];        // K tile  [key][hd]
  __shared__ __align__(16) _Float16 vsh[64][32];        // V tile  [hd][key]
  __shared__ __align__(16) _Float16 ptile[8][16][32];   // wave-private P

  const int tid  = threadIdx.x;
  const int lane = tid & 31;
  const int wave = tid >> 5;
  const int nl = lane & 15, hi = lane >> 4;

  const int bh = blockIdx.y;
  const int b = bh >> 4, h = bh & 15;
  const int Qb = blockIdx.x * 128;                      // block's query base
  const int qw = Qb + wave * 16;                        // wave's query base

  const _Float16* Q = qh + (size_t)bh * T_ * HD_;
  const _Float16* K = kh + (size_t)bh * T_ * HD_;
  const _Float16* V = vt + (size_t)bh * HD_ * T_;

  // Q fragments for this wave's 16 rows (HD=64 -> two K=32 fragments)
  const v16h aq0 = ldfrag_g(Q + (size_t)(qw + nl) * HD_, hi);
  const v16h aq1 = ldfrag_g(Q + (size_t)(qw + nl) * HD_ + 32, hi);

  float mst[8], lst[8];
  v8f y[4] = {};
#pragma unroll
  for (int r = 0; r < 8; ++r) { mst[r] = -1e30f; lst[r] = 0.0f; }

  // cooperative-load index split (done once)
  const int kkey = tid >> 3, koff = (tid & 7) << 3;     // 32 keys x 8 halves
  const int vhd  = tid >> 2, voff = (tid & 3) << 3;     // 64 hds  x 8 halves

  for (int n0 = 0; n0 <= Qb + 127; n0 += 32) {          // uniform over block
    // ---- stage K/V tiles in LDS (each global b128 shared by 8 waves) ----
    *(v8h*)&ksh[kkey][koff] = *(const v8h*)(K + (size_t)(n0 + kkey) * HD_ + koff);
    *(v8h*)&vsh[vhd][voff]  = *(const v8h*)(V + (size_t)vhd * T_ + n0 + voff);
    __syncthreads();

    if (n0 <= qw + 15) {                                // causal range check
      // ---- S = Q K^T (two 16x16 WMMA column groups from LDS) ----
      v8f slo = {}, shi = {};
      {
        v8h l0 = *(const v8h*)&ksh[nl][hi << 3];
        v8h h0 = *(const v8h*)&ksh[nl][(hi << 3) + 16];
        v8h l1 = *(const v8h*)&ksh[nl][32 + (hi << 3)];
        v8h h1 = *(const v8h*)&ksh[nl][32 + (hi << 3) + 16];
        v8h l2 = *(const v8h*)&ksh[nl + 16][hi << 3];
        v8h h2 = *(const v8h*)&ksh[nl + 16][(hi << 3) + 16];
        v8h l3 = *(const v8h*)&ksh[nl + 16][32 + (hi << 3)];
        v8h h3 = *(const v8h*)&ksh[nl + 16][32 + (hi << 3) + 16];
        v16h bk0 = __builtin_shufflevector(l0, h0, 0,1,2,3,4,5,6,7,8,9,10,11,12,13,14,15);
        v16h bk1 = __builtin_shufflevector(l1, h1, 0,1,2,3,4,5,6,7,8,9,10,11,12,13,14,15);
        v16h bk2 = __builtin_shufflevector(l2, h2, 0,1,2,3,4,5,6,7,8,9,10,11,12,13,14,15);
        v16h bk3 = __builtin_shufflevector(l3, h3, 0,1,2,3,4,5,6,7,8,9,10,11,12,13,14,15);
        slo = wmma_f16(aq0, bk0, slo);
        slo = wmma_f16(aq1, bk1, slo);
        shi = wmma_f16(aq0, bk2, shi);
        shi = wmma_f16(aq1, bk3, shi);
      }

      const bool need_mask = (n0 + 31 > qw);
      float plo[8], phi[8], corr[8];
#pragma unroll
      for (int r = 0; r < 8; ++r) {
        float sl = slo[r] * 0.125f;          // 1/sqrt(64)
        float sh = shi[r] * 0.125f;
        if (need_mask) {
          const int qr = qw + r + (hi << 3);
          if (n0 + nl > qr)      sl = -1e30f;
          if (n0 + 16 + nl > qr) sh = -1e30f;
        }
        // row max across the 16 lanes owning this row (C layout half-wave)
        float rm = fmaxf(sl, sh);
        rm = fmaxf(rm, __shfl_xor(rm, 1, 32));
        rm = fmaxf(rm, __shfl_xor(rm, 2, 32));
        rm = fmaxf(rm, __shfl_xor(rm, 4, 32));
        rm = fmaxf(rm, __shfl_xor(rm, 8, 32));
        const float mn = fmaxf(mst[r], rm);
        const float pl = __expf(sl - mn);
        const float ph = __expf(sh - mn);
        float rs = pl + ph;
        rs += __shfl_xor(rs, 1, 32);
        rs += __shfl_xor(rs, 2, 32);
        rs += __shfl_xor(rs, 4, 32);
        rs += __shfl_xor(rs, 8, 32);
        const float co = __expf(mst[r] - mn);
        lst[r] = lst[r] * co + rs;
        mst[r] = mn;
        corr[r] = co;
        plo[r] = pl;
        phi[r] = ph;
      }
#pragma unroll
      for (int j = 0; j < 4; ++j)
#pragma unroll
        for (int r = 0; r < 8; ++r) y[j][r] *= corr[r];

      // ---- transpose P (C layout -> A layout) through wave-private LDS ----
#pragma unroll
      for (int r = 0; r < 8; ++r) {
        ptile[wave][r + (hi << 3)][nl]      = (_Float16)plo[r];
        ptile[wave][r + (hi << 3)][nl + 16] = (_Float16)phi[r];
      }
      asm volatile("s_wait_dscnt 0" ::: "memory");  // intra-wave LDS RAW fence
      v8h pl8 = *(const v8h*)&ptile[wave][nl][hi << 3];
      v8h ph8 = *(const v8h*)&ptile[wave][nl][(hi << 3) + 16];
      v16h pa = __builtin_shufflevector(pl8, ph8,
                                        0,1,2,3,4,5,6,7,8,9,10,11,12,13,14,15);

      // ---- y += P @ V (transposed-V tile in LDS -> contiguous B frags) ----
#pragma unroll
      for (int j = 0; j < 4; ++j) {
        v8h bl = *(const v8h*)&vsh[j * 16 + nl][hi << 3];
        v8h bhh = *(const v8h*)&vsh[j * 16 + nl][(hi << 3) + 16];
        v16h bv = __builtin_shufflevector(bl, bhh,
                                          0,1,2,3,4,5,6,7,8,9,10,11,12,13,14,15);
        y[j] = wmma_f16(pa, bv, y[j]);
      }
    }
    __syncthreads();   // protect LDS tiles before next iteration's stores
  }

  // normalize and store y back in [B,T,D] f16
  float inv[8];
#pragma unroll
  for (int r = 0; r < 8; ++r) inv[r] = 1.0f / lst[r];
#pragma unroll
  for (int j = 0; j < 4; ++j)
#pragma unroll
    for (int r = 0; r < 8; ++r) {
      const int t = qw + r + (hi << 3);
      yh[(size_t)(b * T_ + t) * D_ + h * HD_ + j * 16 + nl] =
          (_Float16)(y[j][r] * inv[r]);
    }
}

// ----------------------------- output projection ---------------------------
__global__ __launch_bounds__(256) void k_gemm_out(
    const _Float16* __restrict__ yh, const _Float16* __restrict__ w2t,
    const float* __restrict__ bias2, float* __restrict__ out) {
  const int lane = threadIdx.x & 31;
  const int wave = threadIdx.x >> 5;
  const int nl = lane & 15, hi = lane >> 4;

  const int n_base = blockIdx.x * 64;                 // 0..1023
  const int m_base = blockIdx.y * 256 + wave * 32;    // 0..4095

  const _Float16* arow0 = yh + (size_t)(m_base + nl) * D_;
  const _Float16* arow1 = yh + (size_t)(m_base + 16 + nl) * D_;
  const _Float16* brow  = w2t + (size_t)(n_base + nl) * D_;

  v8f c[2][4] = {};
  for (int k0 = 0; k0 < D_; k0 += 32) {
    v16h a0 = ldfrag_g(arow0 + k0, hi);
    v16h a1 = ldfrag_g(arow1 + k0, hi);
    __builtin_prefetch(arow0 + k0 + 256, 0, 1);
    __builtin_prefetch(arow1 + k0 + 256, 0, 1);
#pragma unroll
    for (int j = 0; j < 4; ++j) {
      v16h bj = ldfrag_g(brow + (size_t)j * 16 * D_ + k0, hi);
      c[0][j] = wmma_f16(a0, bj, c[0][j]);
      c[1][j] = wmma_f16(a1, bj, c[1][j]);
    }
  }

#pragma unroll
  for (int g = 0; g < 2; ++g) {
#pragma unroll
    for (int j = 0; j < 4; ++j) {
      const int col = n_base + j * 16 + nl;
      const float bv = bias2[col];
#pragma unroll
      for (int r = 0; r < 8; ++r) {
        const int row = m_base + g * 16 + r + (hi << 3);
        out[(size_t)row * D_ + col] = c[g][j][r] + bv;
      }
    }
  }
}

// ------------------------------- host launch -------------------------------
extern "C" void kernel_launch(void* const* d_in, const int* in_sizes, int n_in,
                              void* d_out, int out_size, void* d_ws,
                              size_t ws_size, hipStream_t stream) {
  const float* x  = (const float*)d_in[0];
  const float* W1 = (const float*)d_in[1];
  const float* b1 = (const float*)d_in[2];
  const float* W2 = (const float*)d_in[3];
  const float* b2 = (const float*)d_in[4];
  float* out = (float*)d_out;

  char* ws = (char*)d_ws;
  size_t off = 0;
  auto carve = [&](size_t bytes) {
    char* p = ws + off;
    off = (off + bytes + 255) & ~(size_t)255;
    return p;
  };
  const size_t headElems = (size_t)B_ * H_ * T_ * HD_;
  _Float16* xh  = (_Float16*)carve((size_t)NT_ * D_ * 2);
  _Float16* w1t = (_Float16*)carve((size_t)D3_ * D_ * 2);
  _Float16* w2t = (_Float16*)carve((size_t)D_ * D_ * 2);
  _Float16* qh  = (_Float16*)carve(headElems * 2);
  _Float16* kh  = (_Float16*)carve(headElems * 2);
  _Float16* vt  = (_Float16*)carve(headElems * 2);
  _Float16* yh  = (_Float16*)carve((size_t)NT_ * D_ * 2);
  (void)in_sizes; (void)n_in; (void)out_size; (void)ws_size;

  // precision staging (one-shot, tiny vs GEMM cost)
  k_cvt  <<<(NT_ * D_) / 256, 256, 0, stream>>>(x, xh, NT_ * D_);
  k_cvt_t<<<(D3_ * D_) / 256, 256, 0, stream>>>(W1, w1t, D_, D3_);
  k_cvt_t<<<(D_ * D_) / 256, 256, 0, stream>>>(W2, w2t, D_, D_);

  // QKV projection -> per-head layouts (wave tile 32x64)
  k_gemm_qkv<<<dim3(D3_ / 64, NT_ / 256), 256, 0, stream>>>(xh, w1t, b1,
                                                            qh, kh, vt);
  // causal flash attention with LDS-shared K/V tiles
  k_attn<<<dim3(T_ / 128, B_ * H_), 256, 0, stream>>>(qh, kh, vt, yh);
  // output projection (wave tile 32x64)
  k_gemm_out<<<dim3(D_ / 64, NT_ / 256), 256, 0, stream>>>(yh, w2t, b2, out);
}
